// Attention_86509231276069
// MI455X (gfx1250) — matrix operand order repaired and checked by reading, not compiled, not run on hardware
//
#include <hip/hip_runtime.h>
#include <hip/hip_bf16.h>
#include <math.h>

// ---------------------------------------------------------------------------
// GQA attention (Llama-style) for MI455X / gfx1250.
// All matmuls on v_wmma_f32_16x16x32_bf16 (wave32).
// K-tile staging uses GLOBAL_LOAD_ASYNC_TO_LDS_B128 when the toolchain
// exposes the builtin; synchronous fallback otherwise.
// ---------------------------------------------------------------------------

typedef __attribute__((ext_vector_type(16))) __bf16 v16bf;
typedef __attribute__((ext_vector_type(8)))  __bf16 v8bf;
typedef __attribute__((ext_vector_type(8)))  float  v8f;
typedef __attribute__((ext_vector_type(4)))  int    v4i;

#define BS_   4
#define SEQ_  1024
#define DM_   4096
#define NQ_   32
#define NKV_  8
#define DH_   128
#define DKV_  (NKV_ * DH_)      // 1024
#define E_    (BS_ * SEQ_)      // 4096 rows total

#if defined(__HIP_DEVICE_COMPILE__) && \
    __has_builtin(__builtin_amdgcn_global_load_async_to_lds_b128) && \
    __has_builtin(__builtin_amdgcn_s_wait_asynccnt)
#define HAVE_ASYNC_LDS 1
// builtin signature: (addrspace(1) int4* gsrc, addrspace(3) int4* ldst, imm, imm)
typedef __attribute__((address_space(1))) v4i* gv4ip;
typedef __attribute__((address_space(3))) v4i* lv4ip;
#else
#define HAVE_ASYNC_LDS 0
#endif

// D = A(16x32 bf16) * B(32x16 bf16) + C(16x16 f32)
__device__ __forceinline__ v8f wmma_bf16(v16bf a, v16bf b, v8f c) {
  return __builtin_amdgcn_wmma_f32_16x16x32_bf16(false, a, false, b,
                                                 (short)0, c, false, false);
}

__device__ __forceinline__ v16bf frag16(v8bf lo, v8bf hi) {
  v16bf r;
#pragma unroll
  for (int i = 0; i < 8; ++i) { r[i] = lo[i]; r[i + 8] = hi[i]; }
  return r;
}

// A-fragment (16x32, MxK): lane = {half, m16}; per-lane 16 contiguous elems
// of row m16 at K offset half*16 (matches contiguous B fragments).
__device__ __forceinline__ v16bf load_a_frag(const __bf16* row, int half) {
  v8bf lo = *(const v8bf*)(row + half * 16);
  v8bf hi = *(const v8bf*)(row + half * 16 + 8);
  return frag16(lo, hi);
}

// ---------------------------------------------------------------------------
// f32 -> bf16, 8 elements per thread (n must be a multiple of 8; always is)
// ---------------------------------------------------------------------------
__global__ __launch_bounds__(256) void f32_to_bf16_kernel(const float* __restrict__ in,
                                                          __bf16* __restrict__ out,
                                                          size_t n8) {
  size_t i = (size_t)blockIdx.x * blockDim.x + threadIdx.x;
  if (i >= n8) return;
  const float4* p = (const float4*)in + i * 2;
  float4 a = p[0], b = p[1];
  v8bf o;
  o[0] = (__bf16)a.x; o[1] = (__bf16)a.y; o[2] = (__bf16)a.z; o[3] = (__bf16)a.w;
  o[4] = (__bf16)b.x; o[5] = (__bf16)b.y; o[6] = (__bf16)b.z; o[7] = (__bf16)b.w;
  *(v8bf*)(out + i * 8) = o;
}

// ---------------------------------------------------------------------------
// Tiled GEMM:  C[M,N] = A[M,K] * B[K,N]   (A,B bf16 row-major)
// 256 threads = 8 waves; block tile 128x128; wave tile 32x64; K step 32.
// 8 WMMAs per wave per K-step vs 12 ds_load_b128 (A frags reused over 4 B).
// ---------------------------------------------------------------------------
template <bool BF16OUT>
__global__ __launch_bounds__(256) void gemm_bf16(const __bf16* __restrict__ A,
                                                 const __bf16* __restrict__ B,
                                                 void* __restrict__ Cv,
                                                 int M, int N, int K) {
  __shared__ alignas(32) __bf16 As[128 * 32];   // [m][k]
  __shared__ alignas(32) __bf16 Bt[128 * 32];   // [n][k]  (transposed stage)

  const int tid  = threadIdx.x;
  const int lane = tid & 31;
  const int wid  = tid >> 5;
  const int half = lane >> 4;
  const int m16  = lane & 15;
  const int wm   = wid & 3;    // 4 M-stripes of 32
  const int wn   = wid >> 2;   // 2 N-stripes of 64
  const int m0   = blockIdx.x * 128;
  const int n0   = blockIdx.y * 128;

  v8f acc[2][4];
#pragma unroll
  for (int a = 0; a < 2; ++a)
#pragma unroll
    for (int t = 0; t < 4; ++t) acc[a][t] = (v8f){};

  for (int k0 = 0; k0 < K; k0 += 32) {
    // A tile 128x32: 512 v8bf chunks, 2 per thread
#pragma unroll
    for (int it = 0; it < 2; ++it) {
      int c = tid + it * 256;
      int row = c >> 2, col = (c & 3) * 8;
      *(v8bf*)(As + row * 32 + col) =
          *(const v8bf*)(A + (size_t)(m0 + row) * K + k0 + col);
    }
    // B tile 32x128 -> Bt[n][k]: 64 tasks, each packs 8 k-rows into b128 stores
    if (tid < 64) {
      int kq = tid >> 4;            // 0..3 -> k rows kq*8 .. kq*8+7
      int nc = (tid & 15) * 8;      // 0..120
      v8bf vv[8];
#pragma unroll
      for (int i = 0; i < 8; ++i)
        vv[i] = *(const v8bf*)(B + (size_t)(k0 + kq * 8 + i) * N + n0 + nc);
#pragma unroll
      for (int j = 0; j < 8; ++j) {
        v8bf w;
#pragma unroll
        for (int i = 0; i < 8; ++i) w[i] = vv[i][j];
        *(v8bf*)(Bt + (nc + j) * 32 + kq * 8) = w;
      }
    }
    __syncthreads();

    v16bf af[2];
#pragma unroll
    for (int a = 0; a < 2; ++a)
      af[a] = load_a_frag(As + (wm * 32 + a * 16 + m16) * 32, half);
#pragma unroll
    for (int t = 0; t < 4; ++t) {
      v16bf bf =
          *(const v16bf*)(Bt + (wn * 64 + t * 16 + m16) * 32 + half * 16);
#pragma unroll
      for (int a = 0; a < 2; ++a) acc[a][t] = wmma_bf16(af[a], bf, acc[a][t]);
    }
    __syncthreads();
  }

  // C layout: lane<16 -> M=r, N=lane ; lane>=16 -> M=8+r, N=lane-16
#pragma unroll
  for (int a = 0; a < 2; ++a) {
    const int row_base = m0 + wm * 32 + a * 16 + half * 8;
#pragma unroll
    for (int t = 0; t < 4; ++t) {
      const int col = n0 + wn * 64 + t * 16 + m16;
#pragma unroll
      for (int r = 0; r < 8; ++r) {
        size_t o = (size_t)(row_base + r) * N + col;
        if (BF16OUT) ((__bf16*)Cv)[o] = (__bf16)acc[a][t][r];
        else         ((float*)Cv)[o]  = acc[a][t][r];
      }
    }
  }
}

// ---------------------------------------------------------------------------
// In-place RoPE on bf16 [E_, n_heads*128]; pair (i, i+64) per head.
// ---------------------------------------------------------------------------
__global__ __launch_bounds__(256) void rope_inplace(__bf16* __restrict__ x,
                                                    const int* __restrict__ posp,
                                                    int n_heads, int width,
                                                    int total) {
  int idx = blockIdx.x * blockDim.x + threadIdx.x;
  if (idx >= total) return;
  int i   = idx & 63;
  int t   = idx >> 6;
  int h   = t % n_heads;
  int row = t / n_heads;             // 0..E_-1  (= b*SEQ + s)
  int s   = row & (SEQ_ - 1);
  float pos = (float)(posp[0] + s);
  float inv_freq = __powf(10000.0f, -(float)i * (1.0f / 64.0f));
  float ang = pos * inv_freq;
  float c = cosf(ang), sn = sinf(ang);
  size_t base = (size_t)row * width + (size_t)h * DH_;
  float x1 = (float)x[base + i];
  float x2 = (float)x[base + i + 64];
  x[base + i]      = (__bf16)(x1 * c - x2 * sn);
  x[base + i + 64] = (__bf16)(x2 * c + x1 * sn);
}

// ---------------------------------------------------------------------------
// Flash-style attention.  Block = 128 threads (4 waves); each block handles
// (batch b, q-head h, 64 q rows); each wave a 16-row stripe.
// KV chunk = 64 keys: K[64][128] async-staged to LDS; V^T[128][64] staged via
// register-transpose + ds_store_b128.  S: 16 WMMAs; P->LDS->A-frag; PV: 16.
// ---------------------------------------------------------------------------
__global__ __launch_bounds__(128) void attn_kernel(const __bf16* __restrict__ Q,
                                                   const __bf16* __restrict__ K,
                                                   const __bf16* __restrict__ V,
                                                   const float* __restrict__ mask,
                                                   const int* __restrict__ posp,
                                                   __bf16* __restrict__ O,
                                                   float scale) {
  const int mblk = blockIdx.x;       // 0..15
  const int h    = blockIdx.y;       // 0..31
  const int b    = blockIdx.z;       // 0..3
  const int kvh  = h >> 2;           // 4 q-heads per kv-head
  const int tid  = threadIdx.x;
  const int wid  = tid >> 5;
  const int lane = tid & 31;
  const int half = lane >> 4;
  const int m16  = lane & 15;
  const int pos0 = posp[0];

  __shared__ alignas(32) __bf16 Ks[64 * DH_];     // [key][d]      16 KB
  __shared__ alignas(32) __bf16 Vt[DH_ * 64];     // [d][key]      16 KB
  __shared__ alignas(32) __bf16 Ps[4][16 * 64];   // per-wave P     8 KB

  // Resident Q fragments: 16 rows x 128 d = 4 A-frags
  const int qrow_frag = mblk * 64 + wid * 16 + m16;
  const __bf16* qrowp =
      Q + (size_t)(b * SEQ_ + qrow_frag) * DM_ + (size_t)h * DH_;
  v16bf qf[4];
#pragma unroll
  for (int kk = 0; kk < 4; ++kk) qf[kk] = load_a_frag(qrowp + kk * 32, half);

  const int qrow_c = mblk * 64 + wid * 16 + half * 8;  // + r -> C-layout row

  float m_i[8], l_i[8];
  v8f acc[8];
#pragma unroll
  for (int r = 0; r < 8; ++r) { m_i[r] = -1e30f; l_i[r] = 0.0f; }
#pragma unroll
  for (int t8 = 0; t8 < 8; ++t8) acc[t8] = (v8f){};

  for (int kc = 0; kc < SEQ_ / 64; ++kc) {
    const size_t kvbase =
        (size_t)(b * SEQ_ + kc * 64) * DKV_ + (size_t)kvh * DH_;

    // -------- K chunk -> LDS (async path when available) ----------------
#if HAVE_ASYNC_LDS
#pragma unroll
    for (int it = 0; it < 8; ++it) {
      int c = tid + it * 128;            // 0..1023 -> (key, d-chunk)
      int key = c >> 4;
      int dc  = (c & 15) * 8;
      __builtin_amdgcn_global_load_async_to_lds_b128(
          (gv4ip)(K + kvbase + (size_t)key * DKV_ + dc),
          (lv4ip)(Ks + key * DH_ + dc), 0, 0);
    }
#else
#pragma unroll
    for (int it = 0; it < 8; ++it) {
      int c = tid + it * 128;
      int key = c >> 4;
      int dc  = (c & 15) * 8;
      *(v8bf*)(Ks + key * DH_ + dc) =
          *(const v8bf*)(K + kvbase + (size_t)key * DKV_ + dc);
    }
#endif

    // -------- V chunk -> Vt[d][key]: register transpose, b128 stores ----
    {
      int g  = tid >> 4;           // 8 key-groups of 8
      int dc = (tid & 15) * 8;     // d-chunk
      v8bf vv[8];
#pragma unroll
      for (int i = 0; i < 8; ++i)
        vv[i] = *(const v8bf*)(V + kvbase + (size_t)(g * 8 + i) * DKV_ + dc);
#pragma unroll
      for (int j = 0; j < 8; ++j) {
        v8bf w;
#pragma unroll
        for (int i = 0; i < 8; ++i) w[i] = vv[i][j];
        *(v8bf*)(Vt + (dc + j) * 64 + g * 8) = w;
      }
    }

    // -------- prefetch next chunk (global_prefetch_b8) ------------------
    if (kc + 1 < SEQ_ / 64) {
      size_t nb = kvbase + (size_t)64 * DKV_;
      __builtin_prefetch(K + nb + (size_t)(tid >> 1) * DKV_, 0, 0);
      __builtin_prefetch(V + nb + (size_t)(tid >> 1) * DKV_, 0, 0);
    }

#if HAVE_ASYNC_LDS
    __builtin_amdgcn_s_wait_asynccnt(0);
#endif
    __syncthreads();

    // -------- S = Q * K^T  (4 n-tiles x 4 k-steps) ----------------------
    v8f sacc[4];
#pragma unroll
    for (int nt = 0; nt < 4; ++nt) sacc[nt] = (v8f){};
#pragma unroll
    for (int nt = 0; nt < 4; ++nt)
#pragma unroll
      for (int kk = 0; kk < 4; ++kk) {
        v16bf bf =
            *(const v16bf*)(Ks + (nt * 16 + m16) * DH_ + kk * 32 + half * 16);
        sacc[nt] = wmma_bf16(qf[kk], bf, sacc[nt]);
      }

    // -------- scale + mask, online softmax ------------------------------
    float nm[8];
#pragma unroll
    for (int r = 0; r < 8; ++r) nm[r] = m_i[r];
#pragma unroll
    for (int nt = 0; nt < 4; ++nt)
#pragma unroll
      for (int r = 0; r < 8; ++r) {
        int key = kc * 64 + nt * 16 + m16;
        float sv = sacc[nt][r] * scale +
                   mask[((size_t)b * SEQ_ + (size_t)(pos0 + qrow_c + r)) * SEQ_ + key];
        sacc[nt][r] = sv;
        nm[r] = fmaxf(nm[r], sv);
      }
#pragma unroll
    for (int r = 0; r < 8; ++r) {
      nm[r] = fmaxf(nm[r], __shfl_xor(nm[r], 1, 32));
      nm[r] = fmaxf(nm[r], __shfl_xor(nm[r], 2, 32));
      nm[r] = fmaxf(nm[r], __shfl_xor(nm[r], 4, 32));
      nm[r] = fmaxf(nm[r], __shfl_xor(nm[r], 8, 32));
    }
    float corr[8], rs[8];
#pragma unroll
    for (int r = 0; r < 8; ++r) {
      corr[r] = __expf(m_i[r] - nm[r]);
      m_i[r]  = nm[r];
      rs[r]   = 0.0f;
    }
#pragma unroll
    for (int nt = 0; nt < 4; ++nt)
#pragma unroll
      for (int r = 0; r < 8; ++r) {
        float p = __expf(sacc[nt][r] - nm[r]);
        rs[r] += p;
        Ps[wid][(half * 8 + r) * 64 + nt * 16 + m16] = (__bf16)p;
      }
#pragma unroll
    for (int r = 0; r < 8; ++r) {
      rs[r] += __shfl_xor(rs[r], 1, 32);
      rs[r] += __shfl_xor(rs[r], 2, 32);
      rs[r] += __shfl_xor(rs[r], 4, 32);
      rs[r] += __shfl_xor(rs[r], 8, 32);
      l_i[r] = l_i[r] * corr[r] + rs[r];
    }
#pragma unroll
    for (int t8 = 0; t8 < 8; ++t8)
#pragma unroll
      for (int r = 0; r < 8; ++r) acc[t8][r] *= corr[r];

    __syncthreads();  // make P stores visible / ordered before A-frag reads

    // -------- O += P * V  (re-load P in A-fragment layout) --------------
    v16bf pf[2];
#pragma unroll
    for (int kk2 = 0; kk2 < 2; ++kk2)
      pf[kk2] = load_a_frag(&Ps[wid][m16 * 64 + kk2 * 32], half);
#pragma unroll
    for (int t8 = 0; t8 < 8; ++t8)
#pragma unroll
      for (int kk2 = 0; kk2 < 2; ++kk2) {
        v16bf bf =
            *(const v16bf*)(Vt + (t8 * 16 + m16) * 64 + kk2 * 32 + half * 16);
        acc[t8] = wmma_bf16(pf[kk2], bf, acc[t8]);
      }
    __syncthreads();  // before Ks/Vt are overwritten next chunk
  }

  // -------- finalize: divide by row sums, store bf16 --------------------
  float inv[8];
#pragma unroll
  for (int r = 0; r < 8; ++r) inv[r] = 1.0f / l_i[r];
#pragma unroll
  for (int t8 = 0; t8 < 8; ++t8)
#pragma unroll
    for (int r = 0; r < 8; ++r) {
      size_t o = (size_t)(b * SEQ_ + qrow_c + r) * DM_ +
                 (size_t)h * DH_ + t8 * 16 + m16;
      O[o] = (__bf16)(acc[t8][r] * inv[r]);
    }
}

// ---------------------------------------------------------------------------
// Launch: convert -> QKV GEMMs -> RoPE -> attention -> output GEMM
// ---------------------------------------------------------------------------
extern "C" void kernel_launch(void* const* d_in, const int* in_sizes, int n_in,
                              void* d_out, int out_size, void* d_ws, size_t ws_size,
                              hipStream_t stream) {
  const float* query = (const float*)d_in[0];
  const float* key   = (const float*)d_in[1];
  const float* value = (const float*)d_in[2];
  const float* mask  = (const float*)d_in[3];
  const float* wq    = (const float*)d_in[4];
  const float* wk    = (const float*)d_in[5];
  const float* wv    = (const float*)d_in[6];
  const float* wo    = (const float*)d_in[7];
  const int*   pos   = (const int*)d_in[8];

  char* w = (char*)d_ws;
  size_t off = 0;
  auto alloc = [&](size_t bytes) -> char* {
    char* p = w + off;
    off += (bytes + 255) & ~(size_t)255;
    return p;
  };
  __bf16* xq  = (__bf16*)alloc((size_t)E_ * DM_ * 2);   // bf16 query (reused as attn out)
  __bf16* xk  = (__bf16*)alloc((size_t)E_ * DM_ * 2);
  __bf16* xv  = (__bf16*)alloc((size_t)E_ * DM_ * 2);
  __bf16* wqb = (__bf16*)alloc((size_t)DM_ * DM_ * 2);
  __bf16* wkb = (__bf16*)alloc((size_t)DM_ * DKV_ * 2);
  __bf16* wvb = (__bf16*)alloc((size_t)DM_ * DKV_ * 2);
  __bf16* wob = (__bf16*)alloc((size_t)DM_ * DM_ * 2);
  __bf16* qp  = (__bf16*)alloc((size_t)E_ * DM_ * 2);   // Q proj (bf16, roped in place)
  __bf16* kp  = (__bf16*)alloc((size_t)E_ * DKV_ * 2);
  __bf16* vp  = (__bf16*)alloc((size_t)E_ * DKV_ * 2);
  __bf16* attnb = xq;  // xq dead after Q projection -> reuse for attention output

  auto cvt = [&](const float* src, __bf16* dst, size_t n) {
    size_t n8 = n / 8;
    f32_to_bf16_kernel<<<(unsigned)((n8 + 255) / 256), 256, 0, stream>>>(src, dst, n8);
  };
  cvt(query, xq, (size_t)E_ * DM_);
  cvt(key,   xk, (size_t)E_ * DM_);
  cvt(value, xv, (size_t)E_ * DM_);
  cvt(wq, wqb, (size_t)DM_ * DM_);
  cvt(wk, wkb, (size_t)DM_ * DKV_);
  cvt(wv, wvb, (size_t)DM_ * DKV_);
  cvt(wo, wob, (size_t)DM_ * DM_);

  // Projections (bf16 out), 128x128 block tiles
  gemm_bf16<true><<<dim3(E_ / 128, DM_ / 128), 256, 0, stream>>>(xq, wqb, qp, E_, DM_, DM_);
  gemm_bf16<true><<<dim3(E_ / 128, DKV_ / 128), 256, 0, stream>>>(xk, wkb, kp, E_, DKV_, DM_);
  gemm_bf16<true><<<dim3(E_ / 128, DKV_ / 128), 256, 0, stream>>>(xv, wvb, vp, E_, DKV_, DM_);

  // RoPE in place on Q and K
  {
    int totq = E_ * NQ_ * 64;
    rope_inplace<<<(totq + 255) / 256, 256, 0, stream>>>(qp, pos, NQ_, DM_, totq);
    int totk = E_ * NKV_ * 64;
    rope_inplace<<<(totk + 255) / 256, 256, 0, stream>>>(kp, pos, NKV_, DKV_, totk);
  }

  // Flash attention
  float scale = 1.0f / sqrtf((float)DH_);
  attn_kernel<<<dim3(SEQ_ / 64, NQ_, BS_), 128, 0, stream>>>(qp, kp, vp, mask, pos,
                                                             attnb, scale);

  // Output projection (f32 straight into d_out)
  gemm_bf16<false><<<dim3(E_ / 128, DM_ / 128), 256, 0, stream>>>(attnb, wob, d_out,
                                                                  E_, DM_, DM_);
}
